// VAWEncoder_5308579578355
// MI455X (gfx1250) — matrix operand order repaired and checked by reading, not compiled, not run on hardware
//
#include <hip/hip_runtime.h>

typedef __attribute__((ext_vector_type(2)))  float  v2f;
typedef __attribute__((ext_vector_type(4)))  float  v4f;
typedef __attribute__((ext_vector_type(8)))  float  v8f;
typedef __attribute__((ext_vector_type(8)))  __bf16 v8bf;
typedef __attribute__((ext_vector_type(16))) __bf16 v16bf;

#define B_      32
#define L_      2048
#define E_      256
#define H_      256
#define MAXSTEP 20

// ---------------------------------------------------------------------------
// Weight pre-conversion: cnn[r][k][n] f32  ->  Whi/Wlo[r][n][k] bf16 (split)
// Column-major-in-k layout == exactly the 32x16 bf16 B-fragment feed order.
// ---------------------------------------------------------------------------
__global__ __launch_bounds__(256)
void convert_weights(const float* __restrict__ cnn,
                     __bf16* __restrict__ whi,
                     __bf16* __restrict__ wlo)
{
    const int idx = blockIdx.x * 256 + threadIdx.x;   // [0, 20*256*256)
    const int k = idx & 255;
    const int n = (idx >> 8) & 255;
    const int r = idx >> 16;
    const float  x  = cnn[((size_t)r << 16) | ((size_t)k << 8) | (size_t)n];
    const __bf16 h  = (__bf16)x;
    const __bf16 lo = (__bf16)(x - (float)h);
    whi[idx] = h;
    wlo[idx] = lo;
}

// ---------------------------------------------------------------------------
// Main kernel: 3xBF16 split GEMM on v_wmma_f32_16x16x32_bf16 (XDL pipe).
// Grid: 2048 workgroups (residues 0..7: 103 tiles, 8..19: 102), 32 rows each.
// Wave tile: 32M x 32N (2 msub x 2 nsub), 8 waves tile N = 256.
// ---------------------------------------------------------------------------
__global__ __launch_bounds__(256)
void vaw_encoder_wmma_bf16x3(const float*  __restrict__ word_rep,  // [B,L,E]
                             const __bf16* __restrict__ whi,       // [20,H,E]
                             const __bf16* __restrict__ wlo,       // [20,H,E]
                             float* __restrict__ out,              // [B,L,H]
                             float* __restrict__ out_last)         // [B,H]
{
    __shared__ __bf16 Ahi[32 * E_];   // 16 KB
    __shared__ __bf16 Alo[32 * E_];   // 16 KB

    const int wg = blockIdx.x;
    int r, t;
    if (wg < 8 * 103) { r = wg / 103; t = wg - r * 103; }
    else { int w2 = wg - 8 * 103; int r2 = w2 / 102; r = 8 + r2; t = w2 - r2 * 102; }
    const int n_r = (r < 8) ? 103 : 102;

    const int tid = threadIdx.x;

    // ---- stage + split A tile (32 rows x 256 K) into LDS bf16 hi/lo ----
    {
        const int rr = tid >> 3;               // row 0..31
        const int c0 = tid & 7;                // 32-float slab 0..7
        const int g  = t * 32 + rr;
        const int b  = g / n_r;
        const int j  = g - b * n_r;
        const int l  = r + MAXSTEP * j;
        const v4f* src = (const v4f*)(word_rep + ((size_t)b * L_ + l) * E_) + c0 * 8;
        v4f f[8];
#pragma unroll
        for (int i = 0; i < 8; ++i) f[i] = __builtin_nontemporal_load(&src[i]);
#pragma unroll
        for (int gq = 0; gq < 4; ++gq) {       // 8 floats per group
            v8bf h, lo;
#pragma unroll
            for (int j2 = 0; j2 < 8; ++j2) {
                const float x = f[gq * 2 + (j2 >> 2)][j2 & 3];
                const __bf16 hh = (__bf16)x;
                h[j2]  = hh;
                lo[j2] = (__bf16)(x - (float)hh);
            }
            const int ke = c0 * 32 + gq * 8;
            *(v8bf*)(&Ahi[rr * E_ + ke]) = h;
            *(v8bf*)(&Alo[rr * E_ + ke]) = lo;
        }
    }
    __syncthreads();

    const int lane  = tid & 31;
    const int wv    = tid >> 5;
    const int nbase = wv * 32;
    const int half  = lane >> 4;      // K-half selector within the fragment
    const int lm    = lane & 15;      // M (A) / N (B,C) index

    // A fragment feeds (16-bit A 16x32): lane half 0 -> K {kb..kb+15},
    // half 1 -> K {kb+16..kb+31}; M = lane & 15.
    const __bf16* Ah0 = &Ahi[lm * E_ + half * 16];
    const __bf16* Al0 = &Alo[lm * E_ + half * 16];
    // B fragment feeds (32x16 bf16 B): lanes 0-15 K {kb..kb+15}, lanes 16-31
    // K {kb+16..kb+31}; N = lane & 15. Wcvt[r][n][k] makes this contiguous.
    const size_t nrow0 = ((size_t)(r << 8) + nbase + lm) * E_ + half * 16;
    const __bf16* Bh0 = whi + nrow0;           // nsub 0
    const __bf16* Bl0 = wlo + nrow0;
    const __bf16* Bh1 = Bh0 + 16 * E_;         // nsub 1 (16 columns over)
    const __bf16* Bl1 = Bl0 + 16 * E_;

    v8f acc00 = {}; v8f acc01 = {}; v8f acc10 = {}; v8f acc11 = {};

#pragma unroll 2
    for (int kb = 0; kb < E_; kb += 32) {
        // A fragments (hi/lo, msub 0/1): two 16B runs -> v16bf
        v8bf ah0a = *(const v8bf*)(Ah0 + kb);
        v8bf ah0b = *(const v8bf*)(Ah0 + kb + 8);
        v8bf al0a = *(const v8bf*)(Al0 + kb);
        v8bf al0b = *(const v8bf*)(Al0 + kb + 8);
        v8bf ah1a = *(const v8bf*)(Ah0 + 16 * E_ + kb);
        v8bf ah1b = *(const v8bf*)(Ah0 + 16 * E_ + kb + 8);
        v8bf al1a = *(const v8bf*)(Al0 + 16 * E_ + kb);
        v8bf al1b = *(const v8bf*)(Al0 + 16 * E_ + kb + 8);
        v16bf ah0 = __builtin_shufflevector(ah0a, ah0b, 0,1,2,3,4,5,6,7,8,9,10,11,12,13,14,15);
        v16bf al0 = __builtin_shufflevector(al0a, al0b, 0,1,2,3,4,5,6,7,8,9,10,11,12,13,14,15);
        v16bf ah1 = __builtin_shufflevector(ah1a, ah1b, 0,1,2,3,4,5,6,7,8,9,10,11,12,13,14,15);
        v16bf al1 = __builtin_shufflevector(al1a, al1b, 0,1,2,3,4,5,6,7,8,9,10,11,12,13,14,15);
        // B fragments: 32B contiguous per lane
        v16bf bh0 = *(const v16bf*)(Bh0 + kb);
        v16bf bl0 = *(const v16bf*)(Bl0 + kb);
        v16bf bh1 = *(const v16bf*)(Bh1 + kb);
        v16bf bl1 = *(const v16bf*)(Bl1 + kb);

        // D = A*B + C ; 3xBF16: hi*hi + hi*lo + lo*hi
        acc00 = __builtin_amdgcn_wmma_f32_16x16x32_bf16(false, ah0, false, bl0, (short)0, acc00, false, false);
        acc00 = __builtin_amdgcn_wmma_f32_16x16x32_bf16(false, al0, false, bh0, (short)0, acc00, false, false);
        acc00 = __builtin_amdgcn_wmma_f32_16x16x32_bf16(false, ah0, false, bh0, (short)0, acc00, false, false);

        acc01 = __builtin_amdgcn_wmma_f32_16x16x32_bf16(false, ah0, false, bl1, (short)0, acc01, false, false);
        acc01 = __builtin_amdgcn_wmma_f32_16x16x32_bf16(false, al0, false, bh1, (short)0, acc01, false, false);
        acc01 = __builtin_amdgcn_wmma_f32_16x16x32_bf16(false, ah0, false, bh1, (short)0, acc01, false, false);

        acc10 = __builtin_amdgcn_wmma_f32_16x16x32_bf16(false, ah1, false, bl0, (short)0, acc10, false, false);
        acc10 = __builtin_amdgcn_wmma_f32_16x16x32_bf16(false, al1, false, bh0, (short)0, acc10, false, false);
        acc10 = __builtin_amdgcn_wmma_f32_16x16x32_bf16(false, ah1, false, bh0, (short)0, acc10, false, false);

        acc11 = __builtin_amdgcn_wmma_f32_16x16x32_bf16(false, ah1, false, bl1, (short)0, acc11, false, false);
        acc11 = __builtin_amdgcn_wmma_f32_16x16x32_bf16(false, al1, false, bh1, (short)0, acc11, false, false);
        acc11 = __builtin_amdgcn_wmma_f32_16x16x32_bf16(false, ah1, false, bh1, (short)0, acc11, false, false);
    }

    // ---- store: C VGPR v -> (M = msub*16 + half*8 + v, N = nbase + nsub*16 + lm)
#pragma unroll
    for (int ms = 0; ms < 2; ++ms) {
        const v8f& cv0 = ms ? acc10 : acc00;
        const v8f& cv1 = ms ? acc11 : acc01;
#pragma unroll
        for (int v = 0; v < 8; ++v) {
            const int m = ms * 16 + half * 8 + v;
            const int g = t * 32 + m;
            const int b = g / n_r;
            const int j = g - b * n_r;
            const int l = r + MAXSTEP * j;
            float* orow = out + ((size_t)b * L_ + l) * H_ + nbase + lm;
            __builtin_nontemporal_store(cv0[v], orow);
            __builtin_nontemporal_store(cv1[v], orow + 16);
            if (l == L_ - 1) {
                float* o2 = out_last + (size_t)b * H_ + nbase + lm;
                o2[0]  = cv0[v];
                o2[16] = cv1[v];
            }
        }
    }
}

// ---------------------------------------------------------------------------
// Fallback: full-f32 path on v_wmma_f32_16x16x4_f32 (used if d_ws too small).
// ---------------------------------------------------------------------------
__global__ __launch_bounds__(256)
void vaw_encoder_wmma_f32(const float* __restrict__ word_rep,
                          const float* __restrict__ cnn,
                          float* __restrict__ out,
                          float* __restrict__ out_last)
{
    __shared__ float Alds[32 * E_];

    const int wg = blockIdx.x;
    int r, t;
    if (wg < 8 * 103) { r = wg / 103; t = wg - r * 103; }
    else { int w2 = wg - 8 * 103; int r2 = w2 / 102; r = 8 + r2; t = w2 - r2 * 102; }
    const int n_r = (r < 8) ? 103 : 102;

    const int tid = threadIdx.x;
    {
        const int rr = tid >> 3;
        const int c0 = tid & 7;
        const int g  = t * 32 + rr;
        const int b  = g / n_r;
        const int j  = g - b * n_r;
        const int l  = r + MAXSTEP * j;
        const v4f* src = (const v4f*)(word_rep + ((size_t)b * L_ + l) * E_);
        v4f* dst = (v4f*)(&Alds[rr * E_]);
#pragma unroll
        for (int c = 0; c < 8; ++c)
            dst[c0 + c * 8] = __builtin_nontemporal_load(&src[c0 + c * 8]);
    }
    __syncthreads();

    const int lane  = tid & 31;
    const int wv    = tid >> 5;
    const int nbase = wv * 32;
    const int half  = lane >> 4;
    const int lm    = lane & 15;

    const float* Wr  = cnn + (size_t)r * E_ * H_;
    const float* Bp0 = Wr + (size_t)(half * 2) * H_ + nbase + lm;
    const float* Bp1 = Bp0 + 16;
    const float* Ap0 = &Alds[lm * E_ + half * 2];
    const float* Ap1 = Ap0 + 16 * E_;

    v8f acc00 = {}; v8f acc01 = {}; v8f acc10 = {}; v8f acc11 = {};

#pragma unroll 8
    for (int ks = 0; ks < 64; ++ks) {
        const int ko = ks * 4;
        v2f a0 = *(const v2f*)(Ap0 + ko);
        v2f a1 = *(const v2f*)(Ap1 + ko);
        v2f b0, b1;
        b0.x = Bp0[(size_t)ko * H_];
        b0.y = Bp0[(size_t)(ko + 1) * H_];
        b1.x = Bp1[(size_t)ko * H_];
        b1.y = Bp1[(size_t)(ko + 1) * H_];
        acc00 = __builtin_amdgcn_wmma_f32_16x16x4_f32(false, a0, false, b0, (short)0, acc00, false, false);
        acc01 = __builtin_amdgcn_wmma_f32_16x16x4_f32(false, a0, false, b1, (short)0, acc01, false, false);
        acc10 = __builtin_amdgcn_wmma_f32_16x16x4_f32(false, a1, false, b0, (short)0, acc10, false, false);
        acc11 = __builtin_amdgcn_wmma_f32_16x16x4_f32(false, a1, false, b1, (short)0, acc11, false, false);
    }

#pragma unroll
    for (int ms = 0; ms < 2; ++ms) {
        const v8f& cv0 = ms ? acc10 : acc00;
        const v8f& cv1 = ms ? acc11 : acc01;
#pragma unroll
        for (int v = 0; v < 8; ++v) {
            const int m = ms * 16 + half * 8 + v;
            const int g = t * 32 + m;
            const int b = g / n_r;
            const int j = g - b * n_r;
            const int l = r + MAXSTEP * j;
            float* orow = out + ((size_t)b * L_ + l) * H_ + nbase + lm;
            __builtin_nontemporal_store(cv0[v], orow);
            __builtin_nontemporal_store(cv1[v], orow + 16);
            if (l == L_ - 1) {
                float* o2 = out_last + (size_t)b * H_ + nbase + lm;
                o2[0]  = cv0[v];
                o2[16] = cv1[v];
            }
        }
    }
}

extern "C" void kernel_launch(void* const* d_in, const int* in_sizes, int n_in,
                              void* d_out, int out_size, void* d_ws, size_t ws_size,
                              hipStream_t stream) {
    (void)in_sizes; (void)n_in; (void)out_size;
    const float* word_rep = (const float*)d_in[0];   // [32,2048,256] f32
    const float* cnn      = (const float*)d_in[1];   // [20,256,256]  f32
    float* out      = (float*)d_out;
    float* out_last = (float*)d_out + (size_t)B_ * L_ * H_;

    const size_t welems = (size_t)MAXSTEP * E_ * H_;          // 1,310,720
    const size_t need   = welems * 2 /*hi+lo*/ * 2 /*bytes*/; // 5.25 MB

    if (ws_size >= need) {
        __bf16* whi = (__bf16*)d_ws;
        __bf16* wlo = whi + welems;
        hipLaunchKernelGGL(convert_weights, dim3((unsigned)(welems / 256)), dim3(256),
                           0, stream, cnn, whi, wlo);
        hipLaunchKernelGGL(vaw_encoder_wmma_bf16x3, dim3(2048), dim3(256), 0, stream,
                           word_rep, whi, wlo, out, out_last);
    } else {
        hipLaunchKernelGGL(vaw_encoder_wmma_f32, dim3(2048), dim3(256), 0, stream,
                           word_rep, cnn, out, out_last);
    }
}
